// Mamba3Block_34961033790199
// MI455X (gfx1250) — matrix operand order repaired and checked by reading, not compiled, not run on hardware
//
#include <hip/hip_runtime.h>
#include <hip/hip_bf16.h>

typedef __attribute__((ext_vector_type(8)))  _Float16 v8h;
typedef __attribute__((ext_vector_type(16))) _Float16 v16h;
typedef __attribute__((ext_vector_type(8)))  float    v8f;
typedef int b128_t __attribute__((__vector_size__(16)));  // int4, matches builtin sig

#define DEV_INLINE __device__ __forceinline__

// ---------------------------------------------------------------------------
// CDNA5 async global->LDS helpers (double-buffer pipeline primitives).
// Guarded so the file compiles even if the toolchain lacks the builtins;
// fallback is a synchronous 16B copy through VGPRs.
// ---------------------------------------------------------------------------
DEV_INLINE void async_copy16(const _Float16* g, _Float16* l) {
#if __has_builtin(__builtin_amdgcn_global_load_async_to_lds_b128)
  __builtin_amdgcn_global_load_async_to_lds_b128(
      (__attribute__((address_space(1))) b128_t*)const_cast<_Float16*>(g),
      (__attribute__((address_space(3))) b128_t*)l, 0, 0);
#else
  *(v8h*)l = *(const v8h*)g;
#endif
}
DEV_INLINE void async_wait_le2() {
#if __has_builtin(__builtin_amdgcn_s_wait_asynccnt)
  __builtin_amdgcn_s_wait_asynccnt(2);
#endif
}
DEV_INLINE void async_wait_le0() {
#if __has_builtin(__builtin_amdgcn_s_wait_asynccnt)
  __builtin_amdgcn_s_wait_asynccnt(0);
#endif
}

DEV_INLINE v16h concat8(v8h lo, v8h hi) {
  return __builtin_shufflevector(lo, hi, 0, 1, 2, 3, 4, 5, 6, 7, 8, 9, 10, 11,
                                 12, 13, 14, 15);
}

// Epilogue: mode 0 = bias+residual, 1 = softplus(x+bias), 2 = silu(x)
DEV_INLINE void store_tile(float* __restrict__ C, const v8f& acc, int row0,
                           int col, int ldc, const float* __restrict__ bias,
                           const float* __restrict__ add, int mode) {
  const float bv = bias ? bias[col] : 0.0f;
#pragma unroll
  for (int r = 0; r < 8; ++r) {
    long idx = (long)(row0 + r) * ldc + col;
    float v = acc[r] + bv;
    if (mode == 1) {
      v = (v > 20.f) ? v : log1pf(__expf(v));
    } else if (mode == 2) {
      v = v / (1.f + __expf(-v));
    }
    if (add) v += add[idx];
    C[idx] = v;
  }
}

// ---------------------------------------------------------------------------
// f32 -> f16 elementwise convert
// ---------------------------------------------------------------------------
__global__ void cvt_f32_f16_kernel(const float* __restrict__ in,
                                   _Float16* __restrict__ out, long n) {
  long i = (long)blockIdx.x * blockDim.x + threadIdx.x;
  if (i < n) out[i] = (_Float16)in[i];
}

// ---------------------------------------------------------------------------
// RMSNorm: one block per row of D elements; optional f32 and/or f16 outputs
// ---------------------------------------------------------------------------
__global__ void rmsnorm_kernel(const float* __restrict__ x,
                               const float* __restrict__ w,
                               float* __restrict__ outf,
                               _Float16* __restrict__ outh, int D) {
  __shared__ float red[256];
  long row = blockIdx.x;
  const float* xr = x + row * (long)D;
  float s = 0.f;
  for (int i = threadIdx.x; i < D; i += blockDim.x) {
    float v = xr[i];
    s += v * v;
  }
  red[threadIdx.x] = s;
  __syncthreads();
  for (int off = 128; off > 0; off >>= 1) {
    if ((int)threadIdx.x < off) red[threadIdx.x] += red[threadIdx.x + off];
    __syncthreads();
  }
  float inv = rsqrtf(red[0] / (float)D + 1e-6f);
  for (int i = threadIdx.x; i < D; i += blockDim.x) {
    float v = xr[i] * inv * w[i];
    if (outf) outf[row * (long)D + i] = v;
    if (outh) outh[row * (long)D + i] = (_Float16)v;
  }
}

// ---------------------------------------------------------------------------
// Depthwise causal conv1d (K=4) over L, channels-last (B,L,D); writes f16.
// ---------------------------------------------------------------------------
__global__ void dwconv_kernel(const float* __restrict__ u,
                              const float* __restrict__ kern,   // (D,1,4)
                              const float* __restrict__ bias,   // (D)
                              _Float16* __restrict__ outh,
                              int Bn, int Ln, int Dn) {
  long idx = (long)blockIdx.x * blockDim.x + threadIdx.x;
  long total = (long)Bn * Ln * Dn;
  if (idx >= total) return;
  int c = (int)(idx % Dn);
  long bl = idx / Dn;
  int l = (int)(bl % Ln);
  long b = bl / Ln;
  const float* base = u + b * (long)Ln * Dn;
  float acc = bias[c];
#pragma unroll
  for (int j = 0; j < 4; ++j) {
    int ll = l - 3 + j;
    if (ll >= 0) acc += base[(long)ll * Dn + c] * kern[c * 4 + j];
  }
  outh[idx] = (_Float16)acc;
}

// ---------------------------------------------------------------------------
// LDS-tiled WMMA GEMM with async double buffering.
//   C[M x N] = epilogue( A[M x K] * W[N x K]^T ),  A/W f16 row-major (ld = K)
// Block = 256 threads (8 waves) computing a 64x64 tile; K stepped by 32.
// Per chunk: 256 threads issue one async b128 each for A and W tiles
// (global_load_async_to_lds_b128, ASYNCcnt), s_wait_asynccnt retires the
// compute buffer, waves pull fragments with ds_load_b128, 2 WMMA per wave
// reusing the A fragment across two N-tiles. LDS rows padded to 40 halves
// (80B -> 20-bank stride) to dodge bank conflicts.
// ---------------------------------------------------------------------------
__global__ __launch_bounds__(256) void wmma_gemm_tiled_kernel(
    const _Float16* __restrict__ A, const _Float16* __restrict__ W,
    const float* __restrict__ bias, const float* __restrict__ add,
    float* __restrict__ C, int N, int K, int mode) {
  __shared__ _Float16 Abuf[2][64 * 40];
  __shared__ _Float16 Bbuf[2][64 * 40];

  const int nb = N >> 6;
  const int m0 = (blockIdx.x / nb) << 6;
  const int n0 = (blockIdx.x % nb) << 6;

  const int t = threadIdx.x;
  const int wave = t >> 5, lane = t & 31;
  const int half = lane >> 4, l15 = lane & 15;

  // cooperative copy: thread t owns one 16B chunk of the 64x32 tile
  const int cr = t >> 2;          // row 0..63
  const int cq = (t & 3) << 3;    // half-offset 0,8,16,24
  const _Float16* gA = A + (long)(m0 + cr) * K + cq;
  const _Float16* gW = W + (long)(n0 + cr) * K + cq;
  const int loff = cr * 40 + cq;

  const int wm = wave & 3;          // wave M-tile 0..3
  const int wn = (wave >> 2) << 1;  // wave N-tiles {wn, wn+1}

  v8f acc0 = {}, acc1 = {};
  const int nchunks = K >> 5;

  // prologue: fill buffer 0
  async_copy16(gA, &Abuf[0][loff]);
  async_copy16(gW, &Bbuf[0][loff]);

  for (int i = 0; i < nchunks; ++i) {
    const int b = i & 1;
    if (i + 1 < nchunks) {
      // start filling the other buffer, then retire this buffer's fills
      async_copy16(gA + (i + 1) * 32, &Abuf[1 - b][loff]);
      async_copy16(gW + (i + 1) * 32, &Bbuf[1 - b][loff]);
      async_wait_le2();   // async loads complete in order
    } else {
      async_wait_le0();
    }
    __syncthreads();  // publish LDS fills across waves

    const _Float16* As = &Abuf[b][(wm * 16 + l15) * 40 + half * 8];
    v16h a = concat8(*(const v8h*)As, *(const v8h*)(As + 16));
    const _Float16* Bs0 = &Bbuf[b][(wn * 16 + l15) * 40 + half * 8];
    const _Float16* Bs1 = Bs0 + 16 * 40;
    v16h b0 = concat8(*(const v8h*)Bs0, *(const v8h*)(Bs0 + 16));
    v16h b1 = concat8(*(const v8h*)Bs1, *(const v8h*)(Bs1 + 16));

    acc0 = __builtin_amdgcn_wmma_f32_16x16x32_f16(false, a, false, b0,
                                                  (short)0, acc0, false, false);
    acc1 = __builtin_amdgcn_wmma_f32_16x16x32_f16(false, a, false, b1,
                                                  (short)0, acc1, false, false);
    __syncthreads();  // computed: buffer may be refilled next iteration
  }

  const int rowbase = m0 + wm * 16 + half * 8;
  store_tile(C, acc0, rowbase, n0 + wn * 16 + l15, N, bias, add, mode);
  store_tile(C, acc1, rowbase, n0 + (wn + 1) * 16 + l15, N, bias, add, mode);
}

// ---------------------------------------------------------------------------
// Direct WMMA GEMM (for small / strided-A GEMMs): one wave per 16x16 tile.
// ---------------------------------------------------------------------------
__global__ void wmma_gemm_kernel(const _Float16* __restrict__ A,
                                 const _Float16* __restrict__ W,
                                 const float* __restrict__ bias,
                                 const float* __restrict__ add,
                                 float* __restrict__ C,
                                 int M, int N, int K, int lda, int ldc,
                                 int mode) {
  const int wave = threadIdx.x >> 5;
  const int lane = threadIdx.x & 31;
  const int tilesN = N >> 4;
  const long tilesTotal = (long)(M >> 4) * tilesN;
  long tile = (long)blockIdx.x * (blockDim.x >> 5) + wave;
  if (tile >= tilesTotal) return;
  const int tm = (int)(tile / tilesN);
  const int tn = (int)(tile % tilesN);

  const int half = lane >> 4;
  const int l15 = lane & 15;

  const _Float16* arow = A + (long)(tm * 16 + l15) * lda + half * 8;
  const _Float16* wrow = W + (long)(tn * 16 + l15) * K + half * 8;

  v8f acc = {};
  for (int kk = 0; kk < K; kk += 32) {
    __builtin_prefetch(arow + kk + 128, 0, 1);
    __builtin_prefetch(wrow + kk + 128, 0, 1);
    v8h alo = *(const v8h*)(arow + kk);
    v8h ahi = *(const v8h*)(arow + kk + 16);
    v8h blo = *(const v8h*)(wrow + kk);
    v8h bhi = *(const v8h*)(wrow + kk + 16);
    v16h a = concat8(alo, ahi);
    v16h b = concat8(blo, bhi);
    acc = __builtin_amdgcn_wmma_f32_16x16x32_f16(false, a, false, b, (short)0,
                                                 acc, false, false);
  }
  store_tile(C, acc, tm * 16 + half * 8, tn * 16 + l15, ldc, bias, add, mode);
}

// ---------------------------------------------------------------------------
// Selective scan. One block per (b,h); one thread per d (DH=64 threads).
// 48-element SSM state lives entirely in VGPRs; A/B/C rows staged in LDS.
// ---------------------------------------------------------------------------
__global__ void scan_kernel(const float* __restrict__ dt,    // (B*L*H, 64)
                            const float* __restrict__ uc,    // (B,L,D) f32
                            const float* __restrict__ xp,    // (B*L*H, 160)
                            const float* __restrict__ A_log, // (H, 48)
                            float* __restrict__ y,           // (B,L,D)
                            int Ln) {
  const int H = 16, DH = 64, DS = 48, XPC = 160, DTR = 64;
  const int Dn = H * DH;
  const int bh = blockIdx.x;
  const int b = bh / H, h = bh % H;
  const int d = threadIdx.x;  // 0..63

  __shared__ float As[DS], Bs[DS], Cs[DS];
  if (d < DS) As[d] = -__expf(A_log[h * DS + d]);

  float st[DS];
#pragma unroll
  for (int s = 0; s < DS; ++s) st[s] = 0.f;
  __syncthreads();

  for (int l = 0; l < Ln; ++l) {
    const long blh = ((long)(b * Ln + l)) * H + h;
    const long bld = ((long)(b * Ln + l)) * Dn + h * DH + d;
    const float dtv = dt[blh * DTR + d];
    const float uv = uc[bld];
    __syncthreads();
    if (d < DS) {
      Bs[d] = xp[blh * XPC + DTR + d];
      Cs[d] = xp[blh * XPC + DTR + DS + d];
    }
    __syncthreads();
    const float du = dtv * uv;
    float yv = 0.f;
#pragma unroll
    for (int s = 0; s < DS; ++s) {
      float ab = __expf(dtv * As[s]);
      st[s] = ab * st[s] + du * Bs[s];
      yv += st[s] * Cs[s];
    }
    y[bld] = yv;
  }
}

// ---------------------------------------------------------------------------
// act = (f16)( silu_gate * up )   (gate buffer already holds silu(g))
// ---------------------------------------------------------------------------
__global__ void silu_mul_kernel(const float* __restrict__ g,
                                const float* __restrict__ u,
                                _Float16* __restrict__ out, long n) {
  long i = (long)blockIdx.x * blockDim.x + threadIdx.x;
  if (i < n) out[i] = (_Float16)(g[i] * u[i]);
}

// ---------------------------------------------------------------------------
// Host-side orchestration
// ---------------------------------------------------------------------------
extern "C" void kernel_launch(void* const* d_in, const int* in_sizes, int n_in,
                              void* d_out, int out_size, void* d_ws,
                              size_t ws_size, hipStream_t stream) {
  (void)in_sizes; (void)n_in; (void)out_size; (void)ws_size;

  const int Bn = 4, Ln = 1024, Dn = 1024, Hn = 16, DIn = 2048;
  const int DHn = Dn / Hn;          // 64
  const int DTRn = Dn / 16;         // 64
  const int XPC = DTRn + 2 * 48;    // 160
  const long TOK = (long)Bn * Ln;   // 4096
  const long ROWS = TOK * Hn;       // 65536

  const float* x        = (const float*)d_in[0];
  const float* norm1_w  = (const float*)d_in[1];
  const float* norm2_w  = (const float*)d_in[2];
  const float* dw_k     = (const float*)d_in[3];
  const float* dw_b     = (const float*)d_in[4];
  const float* pw_w     = (const float*)d_in[5];
  const float* pw_b     = (const float*)d_in[6];
  const float* x_proj_w = (const float*)d_in[7];
  const float* dt_proj_w= (const float*)d_in[8];
  const float* dt_proj_b= (const float*)d_in[9];
  const float* A_log    = (const float*)d_in[10];
  const float* out_w    = (const float*)d_in[11];
  const float* out_b    = (const float*)d_in[12];
  const float* gate_w   = (const float*)d_in[13];
  const float* up_w     = (const float*)d_in[14];
  const float* down_w   = (const float*)d_in[15];
  float* out = (float*)d_out;

  char* wp = (char*)d_ws;
  auto carve = [&](size_t bytes) {
    char* p = wp;
    wp += (bytes + 255) & ~(size_t)255;
    return p;
  };
  float*    u_f   = (float*)   carve(TOK * Dn * 4);
  _Float16* convh = (_Float16*)carve(TOK * Dn * 2);
  float*    uc_f  = (float*)   carve(TOK * Dn * 4);
  _Float16* uc_h  = (_Float16*)carve(TOK * Dn * 2);
  float*    xp_f  = (float*)   carve(ROWS * XPC * 4);
  _Float16* xp_h  = (_Float16*)carve(ROWS * XPC * 2);
  float*    dt_f  = (float*)   carve(ROWS * DTRn * 4);
  float*    y_f   = (float*)   carve(TOK * Dn * 4);
  _Float16* y_h   = (_Float16*)carve(TOK * Dn * 2);
  float*    x1_f  = (float*)   carve(TOK * Dn * 4);
  _Float16* h2_h  = (_Float16*)carve(TOK * Dn * 2);
  float*    g_f   = (float*)   carve(TOK * DIn * 4);
  float*    up_f  = (float*)   carve(TOK * DIn * 4);
  _Float16* act_h = (_Float16*)carve(TOK * DIn * 2);
  _Float16* w_pw  = (_Float16*)carve((long)Dn * Dn * 2);
  _Float16* w_xp  = (_Float16*)carve((long)XPC * DHn * 2);
  _Float16* w_dt  = (_Float16*)carve((long)DHn * DTRn * 2);
  _Float16* w_out = (_Float16*)carve((long)Dn * Dn * 2);
  _Float16* w_gate= (_Float16*)carve((long)DIn * Dn * 2);
  _Float16* w_up  = (_Float16*)carve((long)DIn * Dn * 2);
  _Float16* w_down= (_Float16*)carve((long)Dn * DIn * 2);

  auto cvt = [&](const float* src, _Float16* dst, long n) {
    cvt_f32_f16_kernel<<<dim3((unsigned)((n + 255) / 256)), 256, 0, stream>>>(
        src, dst, n);
  };
  // LDS-tiled async GEMM for the big dense projections (lda == K, 64-mult)
  auto gemm_tiled = [&](const _Float16* A, const _Float16* W, const float* bias,
                        const float* add, float* C, int M, int N, int K,
                        int mode) {
    unsigned blocks = (unsigned)((M / 64) * (N / 64));
    wmma_gemm_tiled_kernel<<<dim3(blocks), 256, 0, stream>>>(A, W, bias, add,
                                                             C, N, K, mode);
  };
  // direct GEMM for small / strided-A projections
  auto gemm = [&](const _Float16* A, const _Float16* W, const float* bias,
                  const float* add, float* C, int M, int N, int K, int lda,
                  int ldc, int mode) {
    long tiles = (long)(M / 16) * (N / 16);
    unsigned blocks = (unsigned)((tiles + 7) / 8);
    wmma_gemm_kernel<<<dim3(blocks), 256, 0, stream>>>(A, W, bias, add, C, M,
                                                       N, K, lda, ldc, mode);
  };

  // weight conversions (f32 -> f16)
  cvt(pw_w, w_pw, (long)Dn * Dn);
  cvt(x_proj_w, w_xp, (long)XPC * DHn);
  cvt(dt_proj_w, w_dt, (long)DHn * DTRn);
  cvt(out_w, w_out, (long)Dn * Dn);
  cvt(gate_w, w_gate, (long)DIn * Dn);
  cvt(up_w, w_up, (long)DIn * Dn);
  cvt(down_w, w_down, (long)Dn * DIn);

  // 1) u = rmsnorm(x, norm1_w)
  rmsnorm_kernel<<<dim3((unsigned)TOK), 256, 0, stream>>>(x, norm1_w, u_f,
                                                          nullptr, Dn);
  // 2) depthwise causal conv -> f16
  {
    long n = TOK * Dn;
    dwconv_kernel<<<dim3((unsigned)((n + 255) / 256)), 256, 0, stream>>>(
        u_f, dw_k, dw_b, convh, Bn, Ln, Dn);
  }
  // 3) uc = conv @ pw_w^T + pw_b          (4096 x 1024 x 1024)
  gemm_tiled(convh, w_pw, pw_b, nullptr, uc_f, (int)TOK, Dn, Dn, 0);
  cvt(uc_f, uc_h, TOK * Dn);
  // 4) xp = xh @ x_proj_w^T               (65536 x 64 -> 160)
  gemm(uc_h, w_xp, nullptr, nullptr, xp_f, (int)ROWS, XPC, DHn, DHn, XPC, 0);
  cvt(xp_f, xp_h, ROWS * XPC);
  // 5) dt = softplus(dt_x @ dt_proj_w^T + dt_proj_b)  (A = strided xp slice)
  gemm(xp_h, w_dt, dt_proj_b, nullptr, dt_f, (int)ROWS, DTRn, DTRn, XPC, DTRn,
       1);
  // 6) selective scan -> y
  scan_kernel<<<dim3((unsigned)(Bn * Hn)), DHn, 0, stream>>>(dt_f, uc_f, xp_f,
                                                             A_log, y_f, Ln);
  cvt(y_f, y_h, TOK * Dn);
  // 7) x1 = y @ out_w^T + out_b + residual(x)
  gemm_tiled(y_h, w_out, out_b, x, x1_f, (int)TOK, Dn, Dn, 0);
  // 8) h2 = rmsnorm(x1, norm2_w) -> f16
  rmsnorm_kernel<<<dim3((unsigned)TOK), 256, 0, stream>>>(x1_f, norm2_w,
                                                          nullptr, h2_h, Dn);
  // 9) gate (fused silu) and up GEMMs (4096 x 1024 -> 2048)
  gemm_tiled(h2_h, w_gate, nullptr, nullptr, g_f, (int)TOK, DIn, Dn, 2);
  gemm_tiled(h2_h, w_up, nullptr, nullptr, up_f, (int)TOK, DIn, Dn, 0);
  // 10) act = silu(g) * up  -> f16
  {
    long n = TOK * DIn;
    silu_mul_kernel<<<dim3((unsigned)((n + 255) / 256)), 256, 0, stream>>>(
        g_f, up_f, act_h, n);
  }
  // 11) out = act @ down_w^T + x1         (4096 x 2048 -> 1024)
  gemm_tiled(act_h, w_down, nullptr, x1_f, out, (int)TOK, Dn, DIn, 0);
}
